// DifferentiableRenderer_62234076119163
// MI455X (gfx1250) — matrix operand order repaired and checked by reading, not compiled, not run on hardware
//
#include <hip/hip_runtime.h>
#include <hip/hip_bf16.h>

typedef __attribute__((ext_vector_type(16))) _Float16 v16h;
typedef __attribute__((ext_vector_type(2)))  _Float16 v2h;
typedef __attribute__((ext_vector_type(8)))  float    v8f;

#define HP   128
#define WP   128
#define HWP  (HP * WP)
#define EPSK 1e-8f
#define GSEG 4          // gaussian segments across grid.y

union AHalf {
    v16h v;
    v2h  p[8];
};

__device__ __forceinline__ v2h pack2(float a, float b) {
    // v_cvt_pk_rtz_f16_f32: two f32 -> packed f16 pair
    return __builtin_bit_cast(v2h, __builtin_amdgcn_cvt_pkrtz(a, b));
}

// ---------------------------------------------------------------------------
// Per-gaussian preprocessing.  The RBF exponent is expanded so the inner loop
// is pure FMA:  w = exp2( coef*(x^2+y^2) + bx*x + by*y + c )
//   coef = -0.5*log2(e)/var,  bx = -2*coef*px,  by = -2*coef*py,
//   c    = coef*(px^2+py^2) + log2(opacity)
// gdata[g] = { bx, by, coef, c }
// colh_t   = channel-major f16 colors: colh_t[ch*N + g], ch3 = 1.0 (den col)
// ---------------------------------------------------------------------------
__global__ void splat_prep_kernel(const float* __restrict__ pos,
                                  const float* __restrict__ col,
                                  const float* __restrict__ opac,
                                  const float* __restrict__ scal,
                                  const float* __restrict__ qv,
                                  const float* __restrict__ tv,
                                  float4* __restrict__ gdata,
                                  _Float16* __restrict__ colh_t,
                                  int N) {
    int g = blockIdx.x * blockDim.x + threadIdx.x;
    if (g >= N) return;

    float qw = qv[0], qx = qv[1], qy = qv[2], qz = qv[3];
    float qn = rsqrtf(qw * qw + qx * qx + qy * qy + qz * qz);
    qw *= qn; qx *= qn; qy *= qn; qz *= qn;
    float r00 = 1.f - 2.f * (qy * qy + qz * qz), r01 = 2.f * (qx * qy - qz * qw), r02 = 2.f * (qx * qz + qy * qw);
    float r10 = 2.f * (qx * qy + qz * qw), r11 = 1.f - 2.f * (qx * qx + qz * qz), r12 = 2.f * (qy * qz - qx * qw);
    float r20 = 2.f * (qx * qz - qy * qw), r21 = 2.f * (qy * qz + qx * qw), r22 = 1.f - 2.f * (qx * qx + qy * qy);

    float px = pos[3 * g + 0], py = pos[3 * g + 1], pz = pos[3 * g + 2];
    float cx = r00 * px + r01 * py + r02 * pz + tv[0];
    float cy = r10 * px + r11 * py + r12 * pz + tv[1];
    float cz = r20 * px + r21 * py + r22 * pz + tv[2];
    float iz = 1.0f / cz;
    float sx = cx * iz * 128.0f + 64.0f;   // FX, CX
    float sy = cy * iz * 128.0f + 64.0f;   // FY, CY

    float s    = scal[g];
    float var  = s * s;
    float coef = -0.72134752044448170368f / var;       // -0.5*log2(e)/var
    float lop  = __builtin_amdgcn_logf(opac[g]);       // v_log_f32 == log2
    float bx   = -2.0f * coef * sx;
    float by   = -2.0f * coef * sy;
    float c    = fmaf(coef, fmaf(sx, sx, sy * sy), lop);

    gdata[g] = make_float4(bx, by, coef, c);
    colh_t[0 * N + g] = (_Float16)col[3 * g + 0];
    colh_t[1 * N + g] = (_Float16)col[3 * g + 1];
    colh_t[2 * N + g] = (_Float16)col[3 * g + 2];
    colh_t[3 * N + g] = (_Float16)1.0f;
}

__global__ void splat_zero_kernel(float* __restrict__ accum, int n) {
    int i = blockIdx.x * blockDim.x + threadIdx.x;
    if (i < n) accum[i] = 0.0f;
}

// ---------------------------------------------------------------------------
// Main render: each wave owns 2 pixel tiles of 16 pixels, loops a gaussian
// segment in K=32 chunks.  Per chunk: 3 FMA + 1 v_exp_f32 per (pixel,gauss)
// pair builds two f16 A matrices (16 pix x 32 gauss); B columns are the
// channel-major colors (lane channel = m&3, so no divergence; C cols 4..15
// are never read); two v_wmma_f32_16x16x32_f16 accumulate num/den in f32.
// ---------------------------------------------------------------------------
__global__ __launch_bounds__(256) void splat_render_kernel(
        const float4* __restrict__ gdata,
        const _Float16* __restrict__ colh_t,
        float* __restrict__ accum,
        int N) {
    const int lane = threadIdx.x & 31;
    const int wave = threadIdx.x >> 5;
    const int half = lane >> 4;          // K-subset selector (wave32 layout)
    const int m    = lane & 15;          // A row (pixel-in-tile) / C column

    const int basePix = blockIdx.x * 256 + wave * 32;   // 2 tiles * 16 pixels

    float pxs[2], pys[2], ps2[2];
#pragma unroll
    for (int t = 0; t < 2; ++t) {
        int p = basePix + t * 16 + m;
        pxs[t] = (float)(p & (WP - 1));
        pys[t] = (float)(p >> 7);
        ps2[t] = fmaf(pxs[t], pxs[t], pys[t] * pys[t]);
    }

    const int segLen = N / GSEG;
    const int kstart = blockIdx.y * segLen;

    // Per-lane base pointers: gaussian run gp[kr+e] (K=16*half..16*half+15,
    // contiguous); B elements are 32 contiguous bytes.
    const float4*   gp = gdata + kstart + (half << 4);
    const _Float16* bp = colh_t + (size_t)(m & 3) * N + kstart + (half << 4);

    v8f C0 = {};
    v8f C1 = {};

    for (int kr = 0; kr < segLen; kr += 32) {
        __builtin_prefetch((const void*)(gp + kr + 32), 0, 3);

        // ---- B matrix column (channel m&3) for this lane's K values ----
        AHalf B;
        B.v = *(const v16h*)(bp + kr);

        // ---- A matrices: w = exp2(coef*s2 + bx*x + by*y + c), f16 pack ----
        AHalf A0, A1;
#pragma unroll
        for (int pe = 0; pe < 8; ++pe) {
            const int e0 = 2 * pe, e1 = 2 * pe + 1;
            const int o0 = e0;
            const int o1 = e1;
            const float4 g0 = gp[kr + o0];
            const float4 g1 = gp[kr + o1];

            float w00 = __builtin_amdgcn_exp2f(
                fmaf(g0.z, ps2[0], fmaf(g0.x, pxs[0], fmaf(g0.y, pys[0], g0.w))));
            float w01 = __builtin_amdgcn_exp2f(
                fmaf(g1.z, ps2[0], fmaf(g1.x, pxs[0], fmaf(g1.y, pys[0], g1.w))));
            float w10 = __builtin_amdgcn_exp2f(
                fmaf(g0.z, ps2[1], fmaf(g0.x, pxs[1], fmaf(g0.y, pys[1], g0.w))));
            float w11 = __builtin_amdgcn_exp2f(
                fmaf(g1.z, ps2[1], fmaf(g1.x, pxs[1], fmaf(g1.y, pys[1], g1.w))));

            A0.p[pe] = pack2(w00, w01);
            A1.p[pe] = pack2(w10, w11);
        }

        // ---- accumulate: C += A x B (16x16 f32, cols 0..3 = r,g,b,den) ----
        C0 = __builtin_amdgcn_wmma_f32_16x16x32_f16(false, A0.v, false, B.v,
                                                    (short)0, C0, false, false);
        C1 = __builtin_amdgcn_wmma_f32_16x16x32_f16(false, A1.v, false, B.v,
                                                    (short)0, C1, false, false);
    }

    // ---- combine partial sums across gaussian segments ----
    if (m < 4) {
#pragma unroll
        for (int r = 0; r < 8; ++r) {
            const int row = r + (half << 3);
            atomicAdd(&accum[(size_t)(basePix + row) * 4 + m], C0[r]);
            atomicAdd(&accum[(size_t)(basePix + 16 + row) * 4 + m], C1[r]);
        }
    }
}

// ---------------------------------------------------------------------------
// Finalize: img = num/(den + nchunks*EPS), tile-regrouped output layout.
// ---------------------------------------------------------------------------
__global__ void splat_finalize_kernel(const float* __restrict__ accum,
                                      float* __restrict__ out,
                                      const int* __restrict__ tile_hw_p,
                                      const int* __restrict__ chunk_gauss_p,
                                      int N) {
    int p = blockIdx.x * blockDim.x + threadIdx.x;
    if (p >= HWP) return;
    const int tile = tile_hw_p[0];
    const int step = tile * tile;
    const int nchunks = N / chunk_gauss_p[0];
    float den = accum[p * 4 + 3] + (float)nchunks * EPSK;
    float inv = 1.0f / den;
    const int t = p / step;
    const int s = p - t * step;
    float* ob = out + (size_t)t * 3 * step + s;
    ob[0]        = accum[p * 4 + 0] * inv;
    ob[step]     = accum[p * 4 + 1] * inv;
    ob[2 * step] = accum[p * 4 + 2] * inv;
}

extern "C" void kernel_launch(void* const* d_in, const int* in_sizes, int n_in,
                              void* d_out, int out_size, void* d_ws, size_t ws_size,
                              hipStream_t stream) {
    const float* pos  = (const float*)d_in[0];
    const float* col  = (const float*)d_in[1];
    const float* opac = (const float*)d_in[2];
    const float* scal = (const float*)d_in[3];
    const float* qv   = (const float*)d_in[4];
    const float* tv   = (const float*)d_in[5];
    const int* tile_hw     = (const int*)d_in[6];
    const int* chunk_gauss = (const int*)d_in[7];
    float* out = (float*)d_out;

    const int N = in_sizes[0] / 3;   // 16384

    char* ws = (char*)d_ws;
    float4*   gdata  = (float4*)ws;                              // 16*N bytes
    _Float16* colh_t = (_Float16*)(ws + (size_t)N * 16);         // 8*N bytes
    float*    accum  = (float*)(ws + (size_t)N * 24);            // 16*HWP bytes

    splat_prep_kernel<<<(N + 255) / 256, 256, 0, stream>>>(
        pos, col, opac, scal, qv, tv, gdata, colh_t, N);

    splat_zero_kernel<<<(HWP * 4 + 255) / 256, 256, 0, stream>>>(accum, HWP * 4);

    dim3 grid(HWP / 256, GSEG);   // 64 pixel blocks x 4 gaussian segments
    splat_render_kernel<<<grid, 256, 0, stream>>>(gdata, colh_t, accum, N);

    splat_finalize_kernel<<<(HWP + 255) / 256, 256, 0, stream>>>(
        accum, out, tile_hw, chunk_gauss, N);
}
